// DotProductAttentionTransformerGF_62251255989042
// MI455X (gfx1250) — matrix operand order repaired and checked by reading, not compiled, not run on hardware
//
#include <hip/hip_runtime.h>
#include <hip/hip_bf16.h>
#include <math.h>

typedef _Float16 h16;
typedef __attribute__((ext_vector_type(16))) _Float16 v16h;
typedef __attribute__((ext_vector_type(8)))  _Float16 v8h;
typedef __attribute__((ext_vector_type(8)))  float    v8f;
typedef __attribute__((ext_vector_type(4)))  unsigned int u32x4;
typedef __attribute__((ext_vector_type(8)))  int          i32x8;
typedef __attribute__((ext_vector_type(4)))  int          i32x4;

#define N_NODES 5000
#define N_EDGES 220000L
#define N_BATCH 100
#define S_DIM   256
#define V_DIM   128
#define H_HEADS 8
#define HS_     32
#define HV_     16
#define R_RBF   128
#define FC_     64
#define MS_     768
#define MV_     384
#define F_      512
#define NL_     6
#define CUT_    6.0f
#define AVG_N_  50.0f
#define SQRT3_  1.7320508075688772f
#define INV_SQRT_DEG 0.14997750f   /* 1/sqrt(44.46) */

#if defined(__has_builtin)
# if __has_builtin(__builtin_amdgcn_tensor_load_to_lds)
#  define HAVE_TDM 1
# else
#  define HAVE_TDM 0
# endif
#else
# define HAVE_TDM 0
#endif
#if defined(__has_include)
# if __has_include(<hip/amd_detail/amd_gfx1250_TDM.h>)
#  define TDM_ARGS6 1
# else
#  define TDM_ARGS6 0
# endif
#else
# define TDM_ARGS6 0
#endif

// ---------------------------------------------------------------------------
// TDM: DMA one B tile (32 rows x 64 halfs, row pitch Nc halfs) into LDS with
// 8-half row padding (LDS pitch 72 halfs). D# per CDNA5 ISA ch.8.
// ---------------------------------------------------------------------------
#if HAVE_TDM
__device__ inline void tdm_load_B(const h16* gsrc, unsigned lds_off, int Nc) {
    unsigned long long ga = (unsigned long long)(size_t)gsrc;
    u32x4 g0;
    g0[0] = 1u;                                   // count=1, is_restore=0, gather=0
    g0[1] = lds_off;                              // lds_addr (bytes)
    g0[2] = (unsigned)(ga & 0xFFFFFFFFu);         // global_addr[31:0]
    g0[3] = (unsigned)((ga >> 32) & 0x01FFFFFFu)  // global_addr[56:32]
          | (2u << 30);                           // type=2 (image)
    i32x8 g1;
    // data_size=1 (2B), pad_enable=1, pad_interval=4 (32 dwords = one 64-half
    // row), pad_amount=3 (4 dwords = 8 halfs) -> LDS pitch 72 halfs
    g1[0] = (int)((1u << 16) | (1u << 20) | (4u << 22) | (3u << 25));
    g1[1] = (int)((Nc & 0xFFFF) << 16);           // tensor_dim0 low16 @ bits63:48
    g1[2] = (int)(((Nc >> 16) & 0xFFFF) | (32u << 16)); // dim0 hi | tensor_dim1=32
    g1[3] = (int)(64u << 16);                     // tile_dim0 = 64
    g1[4] = 32;                                   // tile_dim1 = 32, tile_dim2 = 0
    g1[5] = Nc;                                   // tensor_dim0_stride low32
    g1[6] = 0;                                    // stride hi | dim1_stride lo
    g1[7] = 0;
    i32x4 z4 = {0, 0, 0, 0};
#if TDM_ARGS6
    i32x8 z8 = {0, 0, 0, 0, 0, 0, 0, 0};
    __builtin_amdgcn_tensor_load_to_lds(g0, g1, z4, z4, z8, 0);
#else
    __builtin_amdgcn_tensor_load_to_lds(g0, g1, z4, z4, 0);
#endif
}
#endif

// ---------------------------------------------------------------------------
// Generic tiled WMMA GEMM: Y = act(gatherA(X) @ W + bias) [+= residual]
//  Requirements (all call sites satisfy): K % 32 == 0, Nc % 64 == 0.
//  A: f16, row r from rowidx[r] (or r), element (r,k) at
//     A[row*a_row_stride + a_off + k*a_col_stride]
//  W: f16, K x Nc row-major.
//  Block tile 32(M) x 64(N), K-step 32, 128 threads = 4 wave32,
//  each wave -> 16x32 (two v_wmma_f32_16x16x32_f16 per K-step).
//  Double-buffered LDS; B tile staged by the Tensor Data Mover (wave 0),
//  A tile staged with vectorized global->LDS copies.
// ---------------------------------------------------------------------------
__global__ void wmma_gemm_k(const h16* __restrict__ A, const int* __restrict__ rowidx,
                            long a_row_stride, int a_col_stride, long a_off,
                            const h16* __restrict__ W,
                            const float* __restrict__ bias, int act, int accum,
                            float* __restrict__ Yf, h16* __restrict__ Yh,
                            long y_row_stride, int y_col_stride, long y_off,
                            int M, int K, int Nc)
{
    __shared__ __attribute__((aligned(128))) _Float16 As[2][32][40];
    __shared__ __attribute__((aligned(128))) _Float16 Bs[2][32][72];

    const int t    = threadIdx.x;
    const int lane = t & 31;
    const int wave = t >> 5;
    const int m0   = blockIdx.y * 32;
    const int n0   = blockIdx.x * 64;
    const int wm   = (wave >> 1) * 16;   // wave row offset in block tile
    const int wn   = (wave & 1) * 32;    // wave col offset in block tile

    // A staging map: thread -> (row t>>2, 8 halfs at (t&3)*8); M-ragged rows clamp.
    const int ar = t >> 2;
    const int ac = (t & 3) * 8;
    int grow = m0 + ar;
    if (grow >= M) grow = M - 1;              // clamped duplicate; epilogue discards
    const int arow = rowidx ? rowidx[grow] : grow;
    const long abase = (long)arow * a_row_stride + a_off;
    const bool acontig = (a_col_stride == 1);

    auto stage_A = [&](int buf, int k0) {
        v8h v;
        if (acontig) {
            v = *(const v8h*)&A[abase + k0 + ac];
        } else {
            #pragma unroll
            for (int j = 0; j < 8; ++j)
                v[j] = A[abase + (long)(k0 + ac + j) * a_col_stride];
        }
        *(v8h*)&As[buf][ar][ac] = v;
    };
#if !HAVE_TDM
    auto stage_B = [&](int buf, int k0) {
        const int br = t >> 2;
        const int bc = (t & 3) * 16;
        const h16* src = &W[(long)(k0 + br) * Nc + n0 + bc];
        *(v8h*)&Bs[buf][br][bc]     = *(const v8h*)src;
        *(v8h*)&Bs[buf][br][bc + 8] = *(const v8h*)(src + 8);
    };
#endif

    // ---- prologue: stage tile 0 ----
    stage_A(0, 0);
#if HAVE_TDM
    if (wave == 0) {
        tdm_load_B(&W[n0], (unsigned)(size_t)&Bs[0][0][0], Nc);
        __builtin_amdgcn_s_wait_tensorcnt(0);
    }
#else
    stage_B(0, 0);
#endif
    __syncthreads();

    v8f acc0 = {0.f,0.f,0.f,0.f,0.f,0.f,0.f,0.f};
    v8f acc1 = {0.f,0.f,0.f,0.f,0.f,0.f,0.f,0.f};

    for (int k0 = 0; k0 < K; k0 += 32) {
        const int cur = (k0 >> 5) & 1;
        // ---- prefetch next tile into other buffer ----
        if (k0 + 32 < K) {
            if (acontig)
                __builtin_prefetch(&A[abase + k0 + 32 + ac], 0, 1);
            stage_A(cur ^ 1, k0 + 32);
#if HAVE_TDM
            if (wave == 0)
                tdm_load_B(&W[(long)(k0 + 32) * Nc + n0],
                           (unsigned)(size_t)&Bs[cur ^ 1][0][0], Nc);
#else
            stage_B(cur ^ 1, k0 + 32);
#endif
        }
        // ---- fragments (ISA 7.12.2 layouts) + 2x WMMA ----
        const int am  = wm + (lane & 15);
        const int akb = (lane >= 16) ? 8 : 0;
        v8h alo = *(const v8h*)&As[cur][am][akb];
        v8h ahi = *(const v8h*)&As[cur][am][akb + 16];
        const int bn = wn + (lane & 15);
        v16h a, b0, b1;
        #pragma unroll
        for (int i = 0; i < 8; ++i) {
            a[i]  = alo[i];  a[i + 8]  = ahi[i];
            b0[i] = Bs[cur][akb + i][bn]; b0[i + 8] = Bs[cur][akb + 16 + i][bn];
            b1[i] = Bs[cur][akb + i][bn + 16]; b1[i + 8] = Bs[cur][akb + 16 + i][bn + 16];
        }
        acc0 = __builtin_amdgcn_wmma_f32_16x16x32_f16(false, a, false, b0,
                                                      (short)0, acc0, false, false);
        acc1 = __builtin_amdgcn_wmma_f32_16x16x32_f16(false, a, false, b1,
                                                      (short)0, acc1, false, false);
#if HAVE_TDM
        if (wave == 0) __builtin_amdgcn_s_wait_tensorcnt(0);
#endif
        __syncthreads();
    }

    // ---- epilogue: C/D layout -> memory (col always < Nc since Nc%64==0) ----
    const int col0  = n0 + wn + (lane & 15);
    const int rbase = m0 + wm + ((lane >= 16) ? 8 : 0);
    #pragma unroll
    for (int r = 0; r < 8; ++r) {
        const int row = rbase + r;
        if (row < M) {
            #pragma unroll
            for (int half = 0; half < 2; ++half) {
                const int col = col0 + half * 16;
                float val = half ? acc1[r] : acc0[r];
                if (bias) val += bias[col];
                if (act == 1)      val = val / (1.f + __expf(-val));   // SiLU
                else if (act == 2) val = 1.f / (1.f + __expf(-val));   // sigmoid
                const long yi = (long)row * y_row_stride + (long)col * y_col_stride + y_off;
                if (accum) val += Yf[yi];
                if (Yf) Yf[yi] = val;
                if (Yh) Yh[yi] = (h16)val;
            }
        }
    }
}

// ---------------------------------------------------------------------------
// Elementwise / reduction helpers
// ---------------------------------------------------------------------------
__global__ void fill_k(float* p, float v, long n) {
    long i = (long)blockIdx.x * blockDim.x + threadIdx.x;
    if (i < n) p[i] = v;
}
__global__ void cast_k(const float* __restrict__ x, h16* __restrict__ y, long n) {
    long i = (long)blockIdx.x * blockDim.x + threadIdx.x;
    if (i < n) y[i] = (h16)x[i];
}
__global__ void embed_k(const int* __restrict__ z, const float* __restrict__ emb,
                        float* __restrict__ s) {
    long i = (long)blockIdx.x * blockDim.x + threadIdx.x;
    if (i >= (long)N_NODES * S_DIM) return;
    int n = (int)(i / S_DIM), c = (int)(i % S_DIM);
    s[i] = emb[(long)z[n] * S_DIM + c];
}
__global__ void geom_k(const float* __restrict__ pos, const int* __restrict__ src,
                       const int* __restrict__ dst, float* __restrict__ sh1,
                       h16* __restrict__ rbf) {
    long e = (long)blockIdx.x * blockDim.x + threadIdx.x;
    if (e >= N_EDGES) return;
    int s = src[e], d = dst[e];
    float vx = pos[s*3+0]-pos[d*3+0], vy = pos[s*3+1]-pos[d*3+1], vz = pos[s*3+2]-pos[d*3+2];
    float dd = sqrtf(vx*vx + vy*vy + vz*vz);
    float inv = 1.f / (dd + 1e-9f);
    sh1[e*3+0] = SQRT3_*vx*inv; sh1[e*3+1] = SQRT3_*vy*inv; sh1[e*3+2] = SQRT3_*vz*inv;
    const float sig = CUT_ / R_RBF;
    for (int r = 0; r < R_RBF; ++r) {
        float mu = CUT_ * r / (float)(R_RBF - 1);
        float t = (dd - mu) / sig;
        rbf[e*R_RBF + r] = (h16)__expf(-t*t);
    }
}
__global__ void scatter_deg_k(const float* __restrict__ wdeg, const float* __restrict__ sh1,
                              const int* __restrict__ dst, float* __restrict__ s,
                              float* __restrict__ v) {
    long i = (long)blockIdx.x * blockDim.x + threadIdx.x;
    if (i >= N_EDGES * (S_DIM + V_DIM)) return;
    long e = i / (S_DIM + V_DIM);
    int  j = (int)(i % (S_DIM + V_DIM));
    int  d = dst[e];
    float w = wdeg[i] * INV_SQRT_DEG;
    if (j < S_DIM) {
        atomicAdd(&s[(long)d * S_DIM + j], w);
    } else {
        int c = j - S_DIM;
        for (int m = 0; m < 3; ++m)
            atomicAdd(&v[((long)d * V_DIM + c) * 3 + m], w * sh1[e*3+m]);
    }
}
__global__ void ln_k(const float* __restrict__ x, const float* __restrict__ g,
                     const float* __restrict__ b, h16* __restrict__ y, int rows, int L) {
    int row = blockIdx.x * 4 + (threadIdx.x >> 5);
    if (row >= rows) return;
    int lane = threadIdx.x & 31;
    const float* xr = &x[(long)row * L];
    float s = 0.f, s2 = 0.f;
    for (int c = lane; c < L; c += 32) { float t = xr[c]; s += t; s2 += t*t; }
    for (int o = 16; o > 0; o >>= 1) { s += __shfl_xor(s, o, 32); s2 += __shfl_xor(s2, o, 32); }
    float mu = s / L;
    float inv = rsqrtf(s2 / L - mu*mu + 1e-5f);
    for (int c = lane; c < L; c += 32)
        y[(long)row * L + c] = (h16)((xr[c] - mu) * inv * g[c] + b[c]);
}
__global__ void vnorm_k(const float* __restrict__ v, const float* __restrict__ g,
                        h16* __restrict__ y, int rows) {
    int row = blockIdx.x * 4 + (threadIdx.x >> 5);
    if (row >= rows) return;
    int lane = threadIdx.x & 31;
    const float* vr = &v[(long)row * V_DIM * 3];
    float s = 0.f;
    for (int i = lane; i < V_DIM * 3; i += 32) { float t = vr[i]; s += t*t; }
    for (int o = 16; o > 0; o >>= 1) s += __shfl_xor(s, o, 32);
    float inv = rsqrtf(s / V_DIM + 1e-5f);  // 1/sqrt(mean_c(sum_m v^2)+eps)
    for (int i = lane; i < V_DIM * 3; i += 32) {
        int c = i / 3;
        y[(long)row * V_DIM * 3 + i] = (h16)(vr[i] * inv * g[c]);
    }
}
__global__ void svdot_k(const h16* __restrict__ vnh, const int* __restrict__ src,
                        const float* __restrict__ sh1, h16* __restrict__ out) {
    long i = (long)blockIdx.x * blockDim.x + threadIdx.x;
    if (i >= N_EDGES * V_DIM) return;
    long e = i / V_DIM; int c = (int)(i % V_DIM);
    const h16* p = &vnh[((long)src[e] * V_DIM + c) * 3];
    float acc = 0.f;
    for (int m = 0; m < 3; ++m) acc += (float)p[m] * sh1[e*3+m];
    out[i] = (h16)acc;
}
__global__ void combine_ks_k(float* __restrict__ t1 /*in/out -> k_s*/,
                             const float* __restrict__ t2, const float* __restrict__ w) {
    long i = (long)blockIdx.x * blockDim.x + threadIdx.x;
    if (i >= N_EDGES * S_DIM) return;
    long e = i / S_DIM; int c = (int)(i % S_DIM);
    t1[i] = t1[i] * w[e*768 + c] + t2[i] * w[e*768 + 256 + c];
}
__global__ void combine_kv_k(const float* __restrict__ t3, float* __restrict__ t4 /*in/out -> k_v*/,
                             const float* __restrict__ w, const float* __restrict__ sh1) {
    long i = (long)blockIdx.x * blockDim.x + threadIdx.x;
    if (i >= N_EDGES * V_DIM) return;
    long e = i / V_DIM; int c = (int)(i % V_DIM);
    float w3 = w[e*768 + 512 + c], w4 = w[e*768 + 640 + c];
    float a3 = t3[i] * w3;
    for (int m = 0; m < 3; ++m) {
        long idx = i*3 + m;
        t4[idx] = a3 * sh1[e*3+m] + t4[idx] * w4;
    }
}
__device__ inline void atomicMaxF(float* addr, float val) {
    if (val >= 0.f) atomicMax((int*)addr, __float_as_int(val));
    else            atomicMin((unsigned int*)addr, __float_as_uint(val));
}
__global__ void logits_k(const float* __restrict__ qs, const float* __restrict__ qv,
                         const float* __restrict__ ks, const float* __restrict__ kv,
                         const int* __restrict__ dst, float* __restrict__ logits,
                         float* __restrict__ mbuf) {
    long i = (long)blockIdx.x * blockDim.x + threadIdx.x;
    if (i >= N_EDGES * H_HEADS) return;
    long e = i / H_HEADS; int h = (int)(i % H_HEADS); int d = dst[e];
    float acc = 0.f;
    const float* q1 = &qs[(long)d * S_DIM + h * HS_];
    const float* k1 = &ks[e * S_DIM + h * HS_];
    for (int c = 0; c < HS_; ++c) acc += q1[c] * k1[c];
    const float* q2 = &qv[((long)d * V_DIM + h * HV_) * 3];
    const float* k2 = &kv[(e * V_DIM + h * HV_) * 3];
    for (int c = 0; c < HV_ * 3; ++c) acc += q2[c] * k2[c];
    acc *= rsqrtf((float)(HS_ + 3 * HV_));
    logits[i] = acc;
    atomicMaxF(&mbuf[(long)d * H_HEADS + h], acc);
}
__global__ void smexp_k(const float* __restrict__ logits, const int* __restrict__ dst,
                        const float* __restrict__ mbuf, float* __restrict__ a,
                        float* __restrict__ z) {
    long i = (long)blockIdx.x * blockDim.x + threadIdx.x;
    if (i >= N_EDGES * H_HEADS) return;
    long e = i / H_HEADS; int h = (int)(i % H_HEADS); int d = dst[e];
    float mm = mbuf[(long)d * H_HEADS + h];
    if (!__builtin_isfinite(mm)) mm = 0.f;
    float aa = __expf(logits[i] - mm);
    a[i] = aa;
    atomicAdd(&z[(long)d * H_HEADS + h], aa);
}
__global__ void agg_k(const float* __restrict__ a, const float* __restrict__ z,
                      const float* __restrict__ ks, const float* __restrict__ kv,
                      const int* __restrict__ dst, float* __restrict__ os,
                      float* __restrict__ ov) {
    long i = (long)blockIdx.x * blockDim.x + threadIdx.x;
    if (i >= N_EDGES * S_DIM) return;
    long e = i / S_DIM; int c = (int)(i % S_DIM); int d = dst[e];
    int h = c >> 5;
    float an = a[e * H_HEADS + h] / (z[(long)d * H_HEADS + h] + 1e-9f);
    atomicAdd(&os[(long)d * S_DIM + c], an * ks[i]);
    if (c < V_DIM) {
        int h2 = c >> 4;
        float an2 = a[e * H_HEADS + h2] / (z[(long)d * H_HEADS + h2] + 1e-9f);
        for (int m = 0; m < 3; ++m)
            atomicAdd(&ov[((long)d * V_DIM + c) * 3 + m], an2 * kv[(e * V_DIM + c) * 3 + m]);
    }
}
__global__ void gate_k(float* __restrict__ vmid, h16* __restrict__ vmid_h,
                       const float* __restrict__ g, long n /* N*MV */) {
    long i = (long)blockIdx.x * blockDim.x + threadIdx.x;
    if (i >= n) return;
    float gg = g[i];
    for (int m = 0; m < 3; ++m) {
        long idx = i*3 + m;
        float val = vmid[idx] * gg;
        vmid[idx]   = val;
        vmid_h[idx] = (h16)val;
    }
}
__global__ void head_k(const h16* __restrict__ hh, const float* __restrict__ wh2,
                       const float* __restrict__ bh2, const int* __restrict__ batch,
                       float* __restrict__ out) {
    int node = blockIdx.x * 4 + (threadIdx.x >> 5);
    if (node >= N_NODES) return;
    int lane = threadIdx.x & 31;
    float acc = 0.f;
    for (int c = lane; c < F_; c += 32) acc += (float)hh[(long)node * F_ + c] * wh2[c];
    for (int o = 16; o > 0; o >>= 1) acc += __shfl_xor(acc, o, 32);
    if (lane == 0) atomicAdd(&out[batch[node]], (acc + bh2[0]) / AVG_N_);
}

// ---------------------------------------------------------------------------
// Host orchestration
// Param flattening order (jax pytree, sorted dict keys):
//  0 positions 1 atomic_numbers 2 batch 3 edge_src 4 edge_dst
//  5 Wdeg 6 Wh1 7 Wh2 8 atom_embed 9 bh1 10 bh2
//  11.. blocks[0..4] (20 tensors each, sorted):
//    W1s W1v W2s W2v We Wg Wms_s Wms_v Wmv_s Wmv_v Wo_s Wo_v Wq_s Wq_v
//    ln1_b ln1_g ln1_gv ln2_b ln2_g ln2_gv
//  111.. block 5 (16 tensors, sorted):
//    W1s W2s We Wms_s Wms_v Wmv_s Wmv_v Wo_s Wq_s Wq_v Wskip
//    ln1_b ln1_g ln1_gv ln2_b ln2_g
//  127 norm_b 128 norm_g 129 rbf.W0 130 rbf.W1 131 rbf.b0 132 rbf.b1
// ---------------------------------------------------------------------------
extern "C" void kernel_launch(void* const* d_in, const int* in_sizes, int n_in,
                              void* d_out, int out_size, void* d_ws, size_t ws_size,
                              hipStream_t stream) {
    (void)in_sizes; (void)n_in; (void)out_size; (void)ws_size;

    const float* positions = (const float*)d_in[0];
    const int*   atom_z    = (const int*)  d_in[1];
    const int*   batch     = (const int*)  d_in[2];
    const int*   esrc      = (const int*)  d_in[3];
    const int*   edst      = (const int*)  d_in[4];
    float*       out       = (float*)      d_out;

    auto PF = [&](int i) { return (const float*)d_in[i]; };

    size_t off = 0;
    auto alloc = [&](size_t bytes) -> void* {
        void* p = (char*)d_ws + off;
        off = (off + bytes + 255) & ~(size_t)255;
        return p;
    };
    auto launch1d = [&](long n) { return dim3((unsigned)((n + 255) / 256)); };
    auto fillf = [&](float* p, float v, long n) {
        fill_k<<<launch1d(n), 256, 0, stream>>>(p, v, n);
    };
    auto cast16 = [&](const float* x, h16* y, long n) {
        cast_k<<<launch1d(n), 256, 0, stream>>>(x, y, n);
    };
    auto convW = [&](int idx, long n) -> h16* {
        h16* p = (h16*)alloc(n * sizeof(h16));
        cast16(PF(idx), p, n);
        return p;
    };
    auto gemm = [&](const h16* A, const int* ridx, long ars, int acs, long aoff,
                    const h16* W, const float* bias, int act, int accum,
                    float* Yf, h16* Yh, long yrs, int ycs, long yoff,
                    int M, int K, int Nc) {
        dim3 g((Nc + 63) / 64, (M + 31) / 32);
        wmma_gemm_k<<<g, 128, 0, stream>>>(A, ridx, ars, acs, aoff, W, bias, act, accum,
                                           Yf, Yh, yrs, ycs, yoff, M, K, Nc);
    };

    // ---- workspace ----
    float* sh1    = (float*)alloc(N_EDGES * 3 * 4);
    h16*   rbf_h  = (h16*)  alloc(N_EDGES * R_RBF * 2);
    h16*   h1_h   = (h16*)  alloc(N_EDGES * FC_ * 2);
    h16*   f_h    = (h16*)  alloc(N_EDGES * FC_ * 2);
    float* wdeg   = (float*)alloc(N_EDGES * (S_DIM + V_DIM) * 4);
    float* w_e    = (float*)alloc(N_EDGES * 768 * 4);
    float* t1     = (float*)alloc(N_EDGES * S_DIM * 4);     // -> k_s (in place)
    float* t2     = (float*)alloc(N_EDGES * S_DIM * 4);
    float* t3     = (float*)alloc(N_EDGES * V_DIM * 4);
    float* t4     = (float*)alloc(N_EDGES * V_DIM * 3 * 4); // -> k_v (in place)
    float* logit  = (float*)alloc(N_EDGES * H_HEADS * 4);
    float* attn   = (float*)alloc(N_EDGES * H_HEADS * 4);
    h16*   svd_h  = (h16*)  alloc(N_EDGES * V_DIM * 2);

    float* s_buf  = (float*)alloc((long)N_NODES * S_DIM * 4);
    float* v_buf  = (float*)alloc((long)N_NODES * V_DIM * 3 * 4);
    h16*   s_n_h  = (h16*)  alloc((long)N_NODES * S_DIM * 2);
    h16*   v_n_h  = (h16*)  alloc((long)N_NODES * V_DIM * 3 * 2);
    float* q_s    = (float*)alloc((long)N_NODES * S_DIM * 4);
    float* q_v    = (float*)alloc((long)N_NODES * V_DIM * 3 * 4);
    float* mbuf   = (float*)alloc((long)N_NODES * H_HEADS * 4);
    float* zbuf   = (float*)alloc((long)N_NODES * H_HEADS * 4);
    float* o_s    = (float*)alloc((long)N_NODES * S_DIM * 4);
    float* o_v    = (float*)alloc((long)N_NODES * V_DIM * 3 * 4);
    h16*   o_s_h  = (h16*)  alloc((long)N_NODES * S_DIM * 2);
    h16*   o_v_h  = (h16*)  alloc((long)N_NODES * V_DIM * 3 * 2);
    h16*   s_n2_h = (h16*)  alloc((long)N_NODES * F_ * 2);    // fits 256 or 512
    h16*   v_n2_h = (h16*)  alloc((long)N_NODES * V_DIM * 3 * 2);
    h16*   mid_h  = (h16*)  alloc((long)N_NODES * MS_ * 2);
    float* gbuf   = (float*)alloc((long)N_NODES * MV_ * 4);
    float* vmid   = (float*)alloc((long)N_NODES * MV_ * 3 * 4);
    h16*   vmid_h = (h16*)  alloc((long)N_NODES * MV_ * 3 * 2);
    float* s512   = (float*)alloc((long)N_NODES * F_ * 4);
    h16*   h_h    = (h16*)  alloc((long)N_NODES * F_ * 2);

    // ---- setup: geometry, RBF MLP, degree embedding ----
    fillf(out, 0.f, N_BATCH);
    geom_k<<<launch1d(N_EDGES), 256, 0, stream>>>(positions, esrc, edst, sh1, rbf_h);

    h16* W0h   = convW(129, (long)R_RBF * FC_);
    h16* W1h   = convW(130, (long)FC_ * FC_);
    h16* Wdegh = convW(5,   (long)FC_ * (S_DIM + V_DIM));

    gemm(rbf_h, nullptr, R_RBF, 1, 0, W0h, PF(131), 1, 0,
         nullptr, h1_h, FC_, 1, 0, (int)N_EDGES, R_RBF, FC_);
    gemm(h1_h, nullptr, FC_, 1, 0, W1h, PF(132), 1, 0,
         nullptr, f_h, FC_, 1, 0, (int)N_EDGES, FC_, FC_);
    gemm(f_h, nullptr, FC_, 1, 0, Wdegh, nullptr, 0, 0,
         wdeg, nullptr, S_DIM + V_DIM, 1, 0, (int)N_EDGES, FC_, S_DIM + V_DIM);

    embed_k<<<launch1d((long)N_NODES * S_DIM), 256, 0, stream>>>(atom_z, PF(8), s_buf);
    fillf(v_buf, 0.f, (long)N_NODES * V_DIM * 3);
    scatter_deg_k<<<launch1d(N_EDGES * (S_DIM + V_DIM)), 256, 0, stream>>>(
        wdeg, sh1, edst, s_buf, v_buf);

    // ---- transformer blocks ----
    for (int blk = 0; blk < NL_; ++blk) {
        bool last = (blk == NL_ - 1);
        int base = last ? 111 : (11 + 20 * blk);
        int iW1s, iW2s, iWe, iWms_s, iWms_v, iWmv_s, iWmv_v, iWo_s, iWq_s, iWq_v;
        int iW1v = -1, iW2v = -1, iWg = -1, iWo_v = -1, iWskip = -1;
        int iLn1b, iLn1g, iLn1gv, iLn2b, iLn2g, iLn2gv = -1;
        if (!last) {
            iW1s = base+0;  iW1v = base+1;  iW2s = base+2;  iW2v = base+3;
            iWe = base+4;   iWg = base+5;   iWms_s = base+6; iWms_v = base+7;
            iWmv_s = base+8; iWmv_v = base+9; iWo_s = base+10; iWo_v = base+11;
            iWq_s = base+12; iWq_v = base+13;
            iLn1b = base+14; iLn1g = base+15; iLn1gv = base+16;
            iLn2b = base+17; iLn2g = base+18; iLn2gv = base+19;
        } else {
            iW1s = base+0;  iW2s = base+1;  iWe = base+2;
            iWms_s = base+3; iWms_v = base+4; iWmv_s = base+5; iWmv_v = base+6;
            iWo_s = base+7;  iWq_s = base+8; iWq_v = base+9;  iWskip = base+10;
            iLn1b = base+11; iLn1g = base+12; iLn1gv = base+13;
            iLn2b = base+14; iLn2g = base+15;
        }
        int FOUT = last ? F_ : S_DIM;

        h16* Wq_sh  = convW(iWq_s,  (long)S_DIM * S_DIM);
        h16* Wq_vh  = convW(iWq_v,  (long)V_DIM * V_DIM);
        h16* Weh    = convW(iWe,    (long)FC_ * 768);
        h16* Wms_sh = convW(iWms_s, (long)S_DIM * S_DIM);
        h16* Wms_vh = convW(iWms_v, (long)V_DIM * S_DIM);
        h16* Wmv_sh = convW(iWmv_s, (long)S_DIM * V_DIM);
        h16* Wmv_vh = convW(iWmv_v, (long)V_DIM * V_DIM);
        h16* Wo_sh  = convW(iWo_s,  (long)S_DIM * FOUT);
        h16* W1sh   = convW(iW1s,   (long)FOUT * MS_);
        h16* W2sh   = convW(iW2s,   (long)MS_ * FOUT);

        // LN1 / vecnorm
        ln_k<<<(N_NODES + 3) / 4, 128, 0, stream>>>(s_buf, PF(iLn1g), PF(iLn1b), s_n_h, N_NODES, S_DIM);
        vnorm_k<<<(N_NODES + 3) / 4, 128, 0, stream>>>(v_buf, PF(iLn1gv), v_n_h, N_NODES);

        // queries
        gemm(s_n_h, nullptr, S_DIM, 1, 0, Wq_sh, nullptr, 0, 0,
             q_s, nullptr, S_DIM, 1, 0, N_NODES, S_DIM, S_DIM);
        for (int m = 0; m < 3; ++m)
            gemm(v_n_h, nullptr, (long)V_DIM * 3, 3, m, Wq_vh, nullptr, 0, 0,
                 q_v, nullptr, (long)V_DIM * 3, 3, m, N_NODES, V_DIM, V_DIM);

        // edge filters and messages (gather fused into GEMM A loads)
        gemm(f_h, nullptr, FC_, 1, 0, Weh, nullptr, 0, 0,
             w_e, nullptr, 768, 1, 0, (int)N_EDGES, FC_, 768);
        svdot_k<<<launch1d(N_EDGES * V_DIM), 256, 0, stream>>>(v_n_h, esrc, sh1, svd_h);
        gemm(s_n_h, esrc, S_DIM, 1, 0, Wms_sh, nullptr, 0, 0,
             t1, nullptr, S_DIM, 1, 0, (int)N_EDGES, S_DIM, S_DIM);
        gemm(svd_h, nullptr, V_DIM, 1, 0, Wms_vh, nullptr, 0, 0,
             t2, nullptr, S_DIM, 1, 0, (int)N_EDGES, V_DIM, S_DIM);
        gemm(s_n_h, esrc, S_DIM, 1, 0, Wmv_sh, nullptr, 0, 0,
             t3, nullptr, V_DIM, 1, 0, (int)N_EDGES, S_DIM, V_DIM);
        for (int m = 0; m < 3; ++m)
            gemm(v_n_h, esrc, (long)V_DIM * 3, 3, m, Wmv_vh, nullptr, 0, 0,
                 t4, nullptr, (long)V_DIM * 3, 3, m, (int)N_EDGES, V_DIM, V_DIM);
        combine_ks_k<<<launch1d(N_EDGES * S_DIM), 256, 0, stream>>>(t1, t2, w_e);
        combine_kv_k<<<launch1d(N_EDGES * V_DIM), 256, 0, stream>>>(t3, t4, w_e, sh1);

        // segment softmax over dst
        fillf(mbuf, -INFINITY, (long)N_NODES * H_HEADS);
        fillf(zbuf, 0.f, (long)N_NODES * H_HEADS);
        logits_k<<<launch1d(N_EDGES * H_HEADS), 256, 0, stream>>>(q_s, q_v, t1, t4, edst, logit, mbuf);
        smexp_k<<<launch1d(N_EDGES * H_HEADS), 256, 0, stream>>>(logit, edst, mbuf, attn, zbuf);

        // aggregation
        fillf(o_s, 0.f, (long)N_NODES * S_DIM);
        fillf(o_v, 0.f, (long)N_NODES * V_DIM * 3);
        agg_k<<<launch1d(N_EDGES * S_DIM), 256, 0, stream>>>(attn, zbuf, t1, t4, edst, o_s, o_v);
        cast16(o_s, o_s_h, (long)N_NODES * S_DIM);
        cast16(o_v, o_v_h, (long)N_NODES * V_DIM * 3);

        if (!last) {
            h16* Wo_vh = convW(iWo_v, (long)V_DIM * V_DIM);
            h16* Wgh   = convW(iWg,   (long)S_DIM * MV_);
            h16* W1vh  = convW(iW1v,  (long)V_DIM * MV_);
            h16* W2vh  = convW(iW2v,  (long)MV_ * V_DIM);

            // residual: s += o_s @ Wo_s ; v += o_v @ Wo_v (per m)
            gemm(o_s_h, nullptr, S_DIM, 1, 0, Wo_sh, nullptr, 0, 1,
                 s_buf, nullptr, S_DIM, 1, 0, N_NODES, S_DIM, S_DIM);
            for (int m = 0; m < 3; ++m)
                gemm(o_v_h, nullptr, (long)V_DIM * 3, 3, m, Wo_vh, nullptr, 0, 1,
                     v_buf, nullptr, (long)V_DIM * 3, 3, m, N_NODES, V_DIM, V_DIM);

            ln_k<<<(N_NODES + 3) / 4, 128, 0, stream>>>(s_buf, PF(iLn2g), PF(iLn2b), s_n2_h, N_NODES, S_DIM);
            vnorm_k<<<(N_NODES + 3) / 4, 128, 0, stream>>>(v_buf, PF(iLn2gv), v_n2_h, N_NODES);

            gemm(s_n2_h, nullptr, S_DIM, 1, 0, W1sh, nullptr, 1, 0,
                 nullptr, mid_h, MS_, 1, 0, N_NODES, S_DIM, MS_);
            gemm(s_n2_h, nullptr, S_DIM, 1, 0, Wgh, nullptr, 2, 0,
                 gbuf, nullptr, MV_, 1, 0, N_NODES, S_DIM, MV_);
            for (int m = 0; m < 3; ++m)
                gemm(v_n2_h, nullptr, (long)V_DIM * 3, 3, m, W1vh, nullptr, 0, 0,
                     vmid, nullptr, (long)MV_ * 3, 3, m, N_NODES, V_DIM, MV_);
            gate_k<<<launch1d((long)N_NODES * MV_), 256, 0, stream>>>(vmid, vmid_h, gbuf, (long)N_NODES * MV_);
            gemm(mid_h, nullptr, MS_, 1, 0, W2sh, nullptr, 0, 1,
                 s_buf, nullptr, S_DIM, 1, 0, N_NODES, MS_, S_DIM);
            for (int m = 0; m < 3; ++m)
                gemm(vmid_h, nullptr, (long)MV_ * 3, 3, m, W2vh, nullptr, 0, 1,
                     v_buf, nullptr, (long)V_DIM * 3, 3, m, N_NODES, MV_, V_DIM);
        } else {
            h16* Wskiph = convW(iWskip, (long)S_DIM * F_);
            // s512 = o_s @ Wo_s + s_n @ Wskip
            gemm(o_s_h, nullptr, S_DIM, 1, 0, Wo_sh, nullptr, 0, 0,
                 s512, nullptr, F_, 1, 0, N_NODES, S_DIM, F_);
            gemm(s_n_h, nullptr, S_DIM, 1, 0, Wskiph, nullptr, 0, 1,
                 s512, nullptr, F_, 1, 0, N_NODES, S_DIM, F_);
            ln_k<<<(N_NODES + 3) / 4, 128, 0, stream>>>(s512, PF(iLn2g), PF(iLn2b), s_n2_h, N_NODES, F_);
            gemm(s_n2_h, nullptr, F_, 1, 0, W1sh, nullptr, 1, 0,
                 nullptr, mid_h, MS_, 1, 0, N_NODES, F_, MS_);
            gemm(mid_h, nullptr, MS_, 1, 0, W2sh, nullptr, 0, 1,
                 s512, nullptr, F_, 1, 0, N_NODES, MS_, F_);
        }
        (void)iLn2gv;
    }

    // ---- readout head ----
    h16* Wh1h = convW(6, (long)F_ * F_);
    ln_k<<<(N_NODES + 3) / 4, 128, 0, stream>>>(s512, PF(128), PF(127), s_n2_h, N_NODES, F_);
    gemm(s_n2_h, nullptr, F_, 1, 0, Wh1h, PF(9), 1, 0,
         nullptr, h_h, F_, 1, 0, N_NODES, F_, F_);
    head_k<<<(N_NODES + 3) / 4, 128, 0, stream>>>(h_h, PF(7), PF(10), batch, out);
}
